// DrugMPNN_37855841747410
// MI455X (gfx1250) — matrix operand order repaired and checked by reading, not compiled, not run on hardware
//
#include <hip/hip_runtime.h>
#include <hip/hip_bf16.h>
#include <math.h>

typedef _Float16 f16;
typedef __attribute__((ext_vector_type(16))) _Float16 v16h;
typedef __attribute__((ext_vector_type(8)))  _Float16 v8h;
typedef __attribute__((ext_vector_type(8)))  float    v8f;

#define NNODES 131072
#define NEDGES 524288
#define NGRAPH 2048
#define HDIM   128
#define NPG    64      // nodes per graph (N/B, reference batch is contiguous blocks)

// ---------------------------------------------------------------------------
// Weight convert: W [K, Nc] f32 (row-major)  ->  Wt [Nc, Kpad] f16 (transposed,
// zero-padded in K).  Makes B-matrix WMMA fragments contiguous 32B loads.
// ---------------------------------------------------------------------------
__global__ void cvt_w_f16t(const float* __restrict__ W, f16* __restrict__ Wt,
                           int K, int Nc, int Kpad) {
  int idx = blockIdx.x * blockDim.x + threadIdx.x;
  int total = Nc * Kpad;
  if (idx >= total) return;
  int n = idx / Kpad, k = idx - n * Kpad;
  Wt[idx] = (k < K) ? (f16)W[(size_t)k * Nc + n] : (f16)0.0f;
}

// Activation convert: A [M, Kact] f32 -> A16 [M, Kpad] f16, zero-padded in K.
__global__ void cvt_a_f16(const float* __restrict__ A, f16* __restrict__ A16,
                          int M, int Kact, int Kpad) {
  int idx = blockIdx.x * blockDim.x + threadIdx.x;
  int total = M * Kpad;
  if (idx >= total) return;
  int m = idx / Kpad, k = idx - m * Kpad;
  A16[idx] = (k < Kact) ? (f16)A[(size_t)m * Kact + k] : (f16)0.0f;
}

__global__ void zero_u32(unsigned* __restrict__ p, int n) {
  int i = blockIdx.x * blockDim.x + threadIdx.x;
  if (i < n) p[i] = 0u;
}

// ---------------------------------------------------------------------------
// WMMA GEMM:  C[M, Nc](ldc) = op( A16[M, Kpad] @ W[Kpad, Nc] + bias )
//   op: optional inference-BN (x * gamma*rsqrt(1+eps) + beta), optional ReLU.
//   Optional dual-write of an f16 copy of C into C16 (feeds the next GEMM).
// One wave computes a 16x64 output strip (4 accumulators): the A fragment is
// loaded once per K-step (two 16B v8h loads) and reused for 4 v_wmma ops.
// Fragment m/n and C/D layouts per CDNA5 ISA 7.12.2; A/B k order is contiguous (16h + s).
// ---------------------------------------------------------------------------
__global__ __launch_bounds__(256)
void gemm_f16_wmma(const f16* __restrict__ A16, const f16* __restrict__ Wt,
                   const float* __restrict__ bias, const float* __restrict__ gamma,
                   const float* __restrict__ beta, float* __restrict__ C,
                   f16* __restrict__ C16, int M, int Kpad, int Nc, int ldc,
                   int ldc16, int do_relu) {
  const int wpb = blockDim.x >> 5;
  const int t = blockIdx.x * wpb + (threadIdx.x >> 5);
  const int ntn = Nc >> 6;                 // 64-wide output strips
  const int ntiles = (M >> 4) * ntn;
  if (t >= ntiles) return;
  const int tm = t / ntn, tn = t - tm * ntn;
  const int lane = threadIdx.x & 31;
  const int lh = lane >> 4;                // 0 => lanes 0-15, 1 => lanes 16-31
  const int lm = lane & 15;

  const f16* arow  = A16 + (size_t)(tm * 16 + lm) * Kpad + lh * 16;
  const f16* bbase = Wt + (size_t)(tn * 64 + lm) * Kpad + lh * 16;
  const size_t bstride = (size_t)16 * Kpad;

  v8f acc0 = {}, acc1 = {}, acc2 = {}, acc3 = {};
  for (int ks = 0; ks < Kpad; ks += 32) {
    // A k order chosen here (contiguous, same as B): lanes0-15 hold K = ks+[0..15];
    // lanes16-31 hold K = ks+[16..31].  Two aligned 16B loads.
    v8h a0 = *(const v8h*)(arow + ks);
    v8h a1 = *(const v8h*)(arow + ks + 8);
    v16h av;
#pragma unroll
    for (int e = 0; e < 8; ++e) { av[e] = a0[e]; av[e + 8] = a1[e]; }
    // B 32x16: lane = column N, 16 contiguous K halves (lane-half picks block)
    v16h b0 = *(const v16h*)(bbase + ks);
    v16h b1 = *(const v16h*)(bbase + bstride + ks);
    v16h b2 = *(const v16h*)(bbase + 2 * bstride + ks);
    v16h b3 = *(const v16h*)(bbase + 3 * bstride + ks);
    acc0 = __builtin_amdgcn_wmma_f32_16x16x32_f16(false, av, false, b0, (short)0, acc0, false, false);
    acc1 = __builtin_amdgcn_wmma_f32_16x16x32_f16(false, av, false, b1, (short)0, acc1, false, false);
    acc2 = __builtin_amdgcn_wmma_f32_16x16x32_f16(false, av, false, b2, (short)0, acc2, false, false);
    acc3 = __builtin_amdgcn_wmma_f32_16x16x32_f16(false, av, false, b3, (short)0, acc3, false, false);
  }

  const int do_bn = (gamma != nullptr);
  v8f accs[4] = {acc0, acc1, acc2, acc3};
#pragma unroll
  for (int j = 0; j < 4; ++j) {
    const int col = tn * 64 + j * 16 + lm;
    const float bval = bias ? bias[col] : 0.0f;
    float sc = 1.0f, sh = 0.0f;
    if (do_bn) { sc = gamma[col] * rsqrtf(1.0f + 1e-5f); sh = beta[col]; }
#pragma unroll
    for (int r = 0; r < 8; ++r) {
      // C/D layout: VGPR r, lanes 0-15 -> M = r, lanes 16-31 -> M = r + 8
      int row = tm * 16 + r + lh * 8;
      float v = accs[j][r] + bval;
      if (do_bn) v = v * sc + sh;
      if (do_relu) v = fmaxf(v, 0.0f);
      C[(size_t)row * ldc + col] = v;
      if (C16) C16[(size_t)row * ldc16 + col] = (f16)v;
    }
  }
}

// ---------------------------------------------------------------------------
// GCN graph machinery
// ---------------------------------------------------------------------------
__global__ void count_deg(const int* __restrict__ dst, unsigned* __restrict__ cnt,
                          int E) {
  int e = blockIdx.x * blockDim.x + threadIdx.x;
  if (e < E) atomicAdd(&cnt[dst[e]], 1u);
}

__global__ void make_dinv(const unsigned* __restrict__ cnt,
                          float* __restrict__ dinv, int n) {
  int i = blockIdx.x * blockDim.x + threadIdx.x;
  if (i < n) dinv[i] = rsqrtf((float)cnt[i] + 1.0f);   // +1 = self loop
}

// one wave per edge: gather hw[src] (float4/lane), scatter-add into agg[dst]
__global__ __launch_bounds__(256)
void scatter_edges(const int* __restrict__ src, const int* __restrict__ dst,
                   const float* __restrict__ dinv, const float* __restrict__ hw,
                   float* __restrict__ agg, int E) {
  int e = (int)((blockIdx.x * blockDim.x + threadIdx.x) >> 5);
  int lane = threadIdx.x & 31;
  if (e >= E) return;
  int s = src[e], d = dst[e];
  float w = dinv[s] * dinv[d];
  const float4 hv = *(const float4*)(hw + (size_t)s * HDIM + lane * 4);
  float* op = agg + (size_t)d * HDIM + lane * 4;
  __hip_atomic_fetch_add(op + 0, hv.x * w, __ATOMIC_RELAXED, __HIP_MEMORY_SCOPE_AGENT);
  __hip_atomic_fetch_add(op + 1, hv.y * w, __ATOMIC_RELAXED, __HIP_MEMORY_SCOPE_AGENT);
  __hip_atomic_fetch_add(op + 2, hv.z * w, __ATOMIC_RELAXED, __HIP_MEMORY_SCOPE_AGENT);
  __hip_atomic_fetch_add(op + 3, hv.w * w, __ATOMIC_RELAXED, __HIP_MEMORY_SCOPE_AGENT);
}

// h = (residual ? h : 0) + relu(BN(agg + hw*dinv^2 + bias));  also writes the
// f16 mirror h16 consumed directly by the next layer's WMMA GEMM.
__global__ void gcn_epilogue(const float* __restrict__ agg, const float* __restrict__ hw,
                             const float* __restrict__ dinv, const float* __restrict__ bias,
                             const float* __restrict__ gamma, const float* __restrict__ beta,
                             float* __restrict__ h, f16* __restrict__ h16,
                             int residual, int NH) {
  int i = blockIdx.x * blockDim.x + threadIdx.x;
  if (i >= NH) return;
  int v = i >> 7, c = i & (HDIM - 1);
  float di = dinv[v];
  float conv = agg[i] + hw[i] * di * di + bias[c];
  float bn = conv * (gamma[c] * rsqrtf(1.0f + 1e-5f)) + beta[c];
  float r = fmaxf(bn, 0.0f);
  float o = residual ? (h[i] + r) : r;
  h[i] = o;
  h16[i] = (f16)o;
}

// per-graph mean/max pooling over 64 contiguous nodes -> gr[g, 0:128 | 128:256]
__global__ void pool_meanmax(const float* __restrict__ h, float* __restrict__ gr) {
  int g = blockIdx.x;
  int c = threadIdx.x;          // 128 threads, one channel each
  const float* base = h + (size_t)g * NPG * HDIM + c;
  float s = 0.0f, m = -INFINITY;
#pragma unroll 4
  for (int n = 0; n < NPG; ++n) {
    float v = base[(size_t)n * HDIM];
    s += v;
    m = fmaxf(m, v);
  }
  gr[(size_t)g * 256 + c] = s * (1.0f / (float)NPG);
  gr[(size_t)g * 256 + HDIM + c] = m;
}

// out[b] = dot(z2[b, 0:64], Wh3) + bh3
__global__ void head_proj(const float* __restrict__ z2, const float* __restrict__ Wh3,
                          const float* __restrict__ bh3, float* __restrict__ out, int Bn) {
  int b = blockIdx.x * blockDim.x + threadIdx.x;
  if (b >= Bn) return;
  float s = bh3[0];
#pragma unroll
  for (int k = 0; k < 64; ++k) s += z2[(size_t)b * 64 + k] * Wh3[k];
  out[b] = s;
}

// ---------------------------------------------------------------------------
extern "C" void kernel_launch(void* const* d_in, const int* in_sizes, int n_in,
                              void* d_out, int out_size, void* d_ws, size_t ws_size,
                              hipStream_t stream) {
  (void)in_sizes; (void)n_in; (void)out_size; (void)ws_size;
  const float* x        = (const float*)d_in[0];
  const float* gene     = (const float*)d_in[1];
  const int*   eidx     = (const int*)d_in[2];
  const int*   esrc     = eidx;
  const int*   edst     = eidx + NEDGES;
  const float* W_emb    = (const float*)d_in[4];
  const float* b_emb    = (const float*)d_in[5];
  const float* Wg       = (const float*)d_in[6];
  const float* bg       = (const float*)d_in[7];
  const float* bn_gamma = (const float*)d_in[8];
  const float* bn_beta  = (const float*)d_in[9];
  const float* Wd1      = (const float*)d_in[10];
  const float* bd1      = (const float*)d_in[11];
  const float* Wd2      = (const float*)d_in[12];
  const float* bd2      = (const float*)d_in[13];
  const float* Wge1     = (const float*)d_in[14];
  const float* bge1     = (const float*)d_in[15];
  const float* g_gamma  = (const float*)d_in[16];
  const float* g_beta   = (const float*)d_in[17];
  const float* Wge2     = (const float*)d_in[18];
  const float* bge2     = (const float*)d_in[19];
  const float* Wh1      = (const float*)d_in[20];
  const float* bh1      = (const float*)d_in[21];
  const float* h_gamma  = (const float*)d_in[22];
  const float* h_beta   = (const float*)d_in[23];
  const float* Wh2      = (const float*)d_in[24];
  const float* bh2      = (const float*)d_in[25];
  const float* Wh3      = (const float*)d_in[26];
  const float* bh3      = (const float*)d_in[27];
  float* out = (float*)d_out;

  // ---- carve workspace -----------------------------------------------------
  char* wsb = (char*)d_ws;
  size_t off = 0;
  auto carve = [&](size_t bytes) -> char* {
    char* p = wsb + off;
    off = (off + bytes + 255) & ~(size_t)255;
    return p;
  };
  float*    h    = (float*)carve((size_t)NNODES * HDIM * 4);
  float*    hw   = (float*)carve((size_t)NNODES * HDIM * 4);
  float*    agg  = (float*)carve((size_t)NNODES * HDIM * 4);
  unsigned* cnt  = (unsigned*)carve((size_t)NNODES * 4);
  float*    dinv = (float*)carve((size_t)NNODES * 4);
  float*    gr   = (float*)carve((size_t)NGRAPH * 256 * 4);
  float*    d1   = (float*)carve((size_t)NGRAPH * 128 * 4);
  float*    g1   = (float*)carve((size_t)NGRAPH * 256 * 4);
  float*    comb = (float*)carve((size_t)NGRAPH * 256 * 4);
  float*    z1   = (float*)carve((size_t)NGRAPH * 128 * 4);
  float*    z2   = (float*)carve((size_t)NGRAPH * 64 * 4);
  // persistent f16 mirror of h (written by GEMM/epilogue, read by GCN GEMMs)
  f16* h16    = (f16*)carve((size_t)NNODES * HDIM * 2);
  // staging buffer for f32->f16 A conversion (x and the small MLP GEMMs)
  f16* a16    = (f16*)carve((size_t)NGRAPH * 4416 * 2 > (size_t)NNODES * 32 * 2
                            ? (size_t)NGRAPH * 4416 * 2 : (size_t)NNODES * 32 * 2);
  f16* Wt_emb = (f16*)carve((size_t)128 * 32 * 2);
  f16* Wt_g   = (f16*)carve((size_t)3 * 128 * 128 * 2);
  f16* Wt_d1  = (f16*)carve((size_t)128 * 256 * 2);
  f16* Wt_d2  = (f16*)carve((size_t)128 * 128 * 2);
  f16* Wt_ge1 = (f16*)carve((size_t)256 * 4416 * 2);
  f16* Wt_ge2 = (f16*)carve((size_t)128 * 256 * 2);
  f16* Wt_h1  = (f16*)carve((size_t)128 * 256 * 2);
  f16* Wt_h2  = (f16*)carve((size_t)64 * 128 * 2);

  auto cvt = [&](const float* W, f16* Wt, int K, int Nc, int Kpad) {
    int total = Nc * Kpad;
    hipLaunchKernelGGL(cvt_w_f16t, dim3((total + 255) / 256), dim3(256), 0, stream,
                       W, Wt, K, Nc, Kpad);
  };
  // GEMM on an already-f16 A operand
  auto gemm16 = [&](const f16* A16, int Kpad, const f16* Wt, const float* bias,
                    const float* gam, const float* bet, float* C, f16* C16,
                    int M, int Nc, int ldc, int ldc16, int relu) {
    int ntiles = (M / 16) * (Nc / 64);
    hipLaunchKernelGGL(gemm_f16_wmma, dim3((ntiles + 7) / 8), dim3(256), 0, stream,
                       A16, Wt, bias, gam, bet, C, C16, M, Kpad, Nc, ldc, ldc16, relu);
  };
  // stage A (f32 -> padded f16) then GEMM
  auto gemm = [&](const float* A, int Kact, int Kpad, const f16* Wt,
                  const float* bias, const float* gam, const float* bet, float* C,
                  f16* C16, int M, int Nc, int ldc, int ldc16, int relu) {
    int atot = M * Kpad;
    hipLaunchKernelGGL(cvt_a_f16, dim3((atot + 255) / 256), dim3(256), 0, stream,
                       A, a16, M, Kact, Kpad);
    gemm16(a16, Kpad, Wt, bias, gam, bet, C, C16, M, Nc, ldc, ldc16, relu);
  };

  // ---- weight conversion (deterministic, every call) -----------------------
  cvt(W_emb, Wt_emb, 30, 128, 32);
  for (int i = 0; i < 3; ++i)
    cvt(Wg + (size_t)i * 128 * 128, Wt_g + (size_t)i * 128 * 128, 128, 128, 128);
  cvt(Wd1, Wt_d1, 256, 128, 256);
  cvt(Wd2, Wt_d2, 128, 128, 128);
  cvt(Wge1, Wt_ge1, 4415, 256, 4416);
  cvt(Wge2, Wt_ge2, 256, 128, 256);
  cvt(Wh1, Wt_h1, 256, 128, 256);
  cvt(Wh2, Wt_h2, 128, 64, 128);

  // ---- node embedding: h = relu(x @ W_emb + b_emb), dual-write h16 ---------
  gemm(x, 30, 32, Wt_emb, b_emb, nullptr, nullptr, h, h16,
       NNODES, 128, 128, 128, 1);

  // ---- degrees / symmetric normalization -----------------------------------
  hipLaunchKernelGGL(zero_u32, dim3(NNODES / 256), dim3(256), 0, stream, cnt, NNODES);
  hipLaunchKernelGGL(count_deg, dim3(NEDGES / 256), dim3(256), 0, stream,
                     edst, cnt, NEDGES);
  hipLaunchKernelGGL(make_dinv, dim3(NNODES / 256), dim3(256), 0, stream,
                     cnt, dinv, NNODES);

  // ---- 3 GCN layers (GEMM A comes straight from the f16 mirror) ------------
  const int NH = NNODES * HDIM;
  for (int i = 0; i < 3; ++i) {
    gemm16(h16, 128, Wt_g + (size_t)i * 128 * 128, nullptr, nullptr, nullptr,
           hw, nullptr, NNODES, 128, 128, 0, 0);
    hipLaunchKernelGGL(zero_u32, dim3(NH / 256), dim3(256), 0, stream,
                       (unsigned*)agg, NH);
    hipLaunchKernelGGL(scatter_edges, dim3(NEDGES * 32 / 256), dim3(256), 0, stream,
                       esrc, edst, dinv, hw, agg, NEDGES);
    hipLaunchKernelGGL(gcn_epilogue, dim3(NH / 256), dim3(256), 0, stream,
                       agg, hw, dinv, bg + (size_t)i * 128,
                       bn_gamma + (size_t)i * 128, bn_beta + (size_t)i * 128,
                       h, h16, (i > 0) ? 1 : 0, NH);
  }

  // ---- pooling -> graph_repr [B, 256] --------------------------------------
  hipLaunchKernelGGL(pool_meanmax, dim3(NGRAPH), dim3(128), 0, stream, h, gr);

  // ---- drug branch: comb[:,0:128] = relu(gr@Wd1+bd1) @ Wd2 + bd2 -----------
  gemm(gr, 256, 256, Wt_d1, bd1, nullptr, nullptr, d1, nullptr,
       NGRAPH, 128, 128, 0, 1);
  gemm(d1, 128, 128, Wt_d2, bd2, nullptr, nullptr, comb, nullptr,
       NGRAPH, 128, 256, 0, 0);

  // ---- gene branch: comb[:,128:256] = relu(relu(BN(gene@Wge1+bge1))@Wge2+bge2)
  gemm(gene, 4415, 4416, Wt_ge1, bge1, g_gamma, g_beta, g1, nullptr,
       NGRAPH, 256, 256, 0, 1);
  gemm(g1, 256, 256, Wt_ge2, bge2, nullptr, nullptr, comb + 128, nullptr,
       NGRAPH, 128, 256, 0, 1);

  // ---- head ----------------------------------------------------------------
  gemm(comb, 256, 256, Wt_h1, bh1, h_gamma, h_beta, z1, nullptr,
       NGRAPH, 128, 128, 0, 1);
  gemm(z1, 128, 128, Wt_h2, bh2, nullptr, nullptr, z2, nullptr,
       NGRAPH, 64, 64, 0, 1);
  hipLaunchKernelGGL(head_proj, dim3(NGRAPH / 256), dim3(256), 0, stream,
                     z2, Wh3, bh3, out, NGRAPH);
}
